// DecoderLayer_66408784331382
// MI455X (gfx1250) — matrix operand order repaired and checked by reading, not compiled, not run on hardware
//
#include <hip/hip_runtime.h>

// ---------------------------------------------------------------------------
// Decoder layer for MI455X (gfx1250, wave32, WMMA 16x16x32 bf16).
// GEMMs run on v_wmma_f32_16x16x32_bf16 with a double-buffered Tensor-Data-
// Mover (TDM) pipeline: one wave issues tensor_load_to_lds descriptors (with
// hardware LDS padding matching the 40-element fragment stride) while all 8
// waves compute WMMAs on the other buffer; sync via s_wait_tensorcnt+barrier.
// fp32 weights are converted+transposed to bf16 [N][K] into one reused slot.
// ---------------------------------------------------------------------------

#define SQ   2048   // sequence length
#define HIDD 2048   // hidden size
#define NHQ  32     // query heads
#define NKVH 8      // kv heads
#define HDIM 64     // head dim
#define INTR 8192   // mlp intermediate

typedef __attribute__((ext_vector_type(16))) __bf16 v16bf;
typedef __attribute__((ext_vector_type(8)))  float  v8f;
typedef __attribute__((ext_vector_type(4)))  unsigned int u32x4;
typedef __attribute__((ext_vector_type(8)))  int i32x8;
typedef __attribute__((ext_vector_type(4)))  int i32x4;

#if __has_builtin(__builtin_amdgcn_tensor_load_to_lds) && \
    __has_builtin(__builtin_amdgcn_s_wait_tensorcnt)
#define DL_TDM 1
#endif

__device__ __forceinline__ v8f wmma_bf16(v16bf a, v16bf b, v8f c) {
  // (neg_a, A, neg_b, B, c_mod, C, reuse_a, reuse_b)
  return __builtin_amdgcn_wmma_f32_16x16x32_bf16(false, a, false, b, (short)0, c,
                                                 false, false);
}

// A-fragment (16x32 bf16, MxK): lanes 0-15 -> M=0-15, lane half selects K-group.
// VGPR v: K = 2v + 16*half.
__device__ __forceinline__ v16bf load_a_frag(const __bf16* lds, int row_base,
                                             int k_off, int stride) {
  int lane = threadIdx.x & 31;
  int m = lane & 15, hl = lane >> 4;
  const __bf16* p = lds + (row_base + m) * stride + k_off + 16 * hl;
  v16bf f;
#pragma unroll
  for (int v = 0; v < 8; ++v) { f[2*v] = p[2*v]; f[2*v+1] = p[2*v+1]; }
  return f;
}

// B-fragment (32x16 bf16, KxN): lane n = lane&15, VGPR v: K = 2v + 16*half.
// LDS layout must be [n][k] (k contiguous).
__device__ __forceinline__ v16bf load_b_frag(const __bf16* lds, int n_base,
                                             int k_off, int stride) {
  int lane = threadIdx.x & 31;
  int n = lane & 15, hl = lane >> 4;
  const __bf16* p = lds + (n_base + n) * stride + k_off + 16 * hl;
  v16bf f;
#pragma unroll
  for (int v = 0; v < 8; ++v) { f[2*v] = p[2*v]; f[2*v+1] = p[2*v+1]; }
  return f;
}

__device__ __forceinline__ float half_reduce_max(float v) {
  v = fmaxf(v, __shfl_xor(v, 1)); v = fmaxf(v, __shfl_xor(v, 2));
  v = fmaxf(v, __shfl_xor(v, 4)); v = fmaxf(v, __shfl_xor(v, 8));
  return v;  // xor<=8 stays inside each 16-lane half (one matrix row)
}
__device__ __forceinline__ float half_reduce_sum(float v) {
  v += __shfl_xor(v, 1); v += __shfl_xor(v, 2);
  v += __shfl_xor(v, 4); v += __shfl_xor(v, 8);
  return v;
}

#ifdef DL_TDM
// Issue a 2-D TDM tile load: tile_x (bf16 elems) x tile_y rows, row stride
// stride_x elems, from global tile-start gaddr into LDS at lds_addr.
// Hardware padding: after every 16 DWORDs (one 32-elem bf16 row = 64B) insert
// 4 DWORDs (16B = 8 elems) -> LDS row stride 40 elems == LDSA_STRIDE.
__device__ __forceinline__ void tdm_load_2d(unsigned lds_addr, const void* gaddr,
                                            unsigned tile_x, unsigned tile_y,
                                            unsigned tensor_x, unsigned tensor_y,
                                            unsigned stride_x) {
  unsigned long long ga = (unsigned long long)(uintptr_t)gaddr;
  u32x4 g0;
  g0[0] = 1u;                                   // count=1, user descriptor
  g0[1] = lds_addr;                             // LDS byte address
  g0[2] = (unsigned)(ga & 0xffffffffu);         // global addr [31:0]
  g0[3] = (unsigned)((ga >> 32) & 0x1ffffffu)   // global addr [56:32]
          | (2u << 30);                         // type = 2 ("image")
  i32x8 g1;
  g1[0] = (1 << 16)      // data_size = 1 (2 bytes)
        | (1 << 20)      // pad_enable
        | (3 << 22)      // pad_interval = 16 DWORDs
        | (3 << 25);     // pad_amount   = 4 DWORDs
  g1[1] = (int)((tensor_x & 0xffffu) << 16);                    // dim0[15:0]
  g1[2] = (int)((tensor_x >> 16) & 0xffffu)                     // dim0[31:16]
        | (int)((tensor_y & 0xffffu) << 16);                    // dim1[15:0]
  g1[3] = (int)((tensor_y >> 16) & 0xffffu)                     // dim1[31:16]
        | (int)((tile_x & 0xffffu) << 16);                      // tile_dim0
  g1[4] = (int)(tile_y & 0xffffu);                              // tile_dim1
  g1[5] = (int)stride_x;                                        // dim0 stride lo
  g1[6] = 0;                                                    // stride hi / dim1 stride lo
  g1[7] = 0;
  i32x4 z4 = {0, 0, 0, 0};
  i32x8 z8 = {0, 0, 0, 0, 0, 0, 0, 0};
  // clang-23 / therock-10.0 form: (g0, g1, g2, g3, x8, cpol)
  __builtin_amdgcn_tensor_load_to_lds(g0, g1, z4, z4, z8, 0);
}
#endif

// ---------------------------------------------------------------------------
// RMSNorm: fp32 [S][HID] -> bf16 [S][HID]
// ---------------------------------------------------------------------------
__global__ __launch_bounds__(256) void dl_rmsnorm_kernel(
    const float* __restrict__ x, const float* __restrict__ w,
    __bf16* __restrict__ out) {
  int row = blockIdx.x;
  const float* xr = x + (size_t)row * HIDD;
  __shared__ float red[256];
  float ss = 0.f;
  for (int c = threadIdx.x; c < HIDD; c += 256) { float v = xr[c]; ss += v * v; }
  red[threadIdx.x] = ss;
  __syncthreads();
  for (int off = 128; off > 0; off >>= 1) {
    if (threadIdx.x < off) red[threadIdx.x] += red[threadIdx.x + off];
    __syncthreads();
  }
  float inv = rsqrtf(red[0] * (1.0f / HIDD) + 1e-6f);
  for (int c = threadIdx.x; c < HIDD; c += 256)
    out[(size_t)row * HIDD + c] = (__bf16)(xr[c] * inv * w[c]);
}

// ---------------------------------------------------------------------------
// Weight convert + transpose: fp32 src[K][N] -> bf16 dst[N][K]
// ---------------------------------------------------------------------------
__global__ __launch_bounds__(256) void dl_convt_kernel(
    const float* __restrict__ src, __bf16* __restrict__ dst, int K, int N) {
  __shared__ __bf16 t[32][33];
  int n0 = blockIdx.x * 32, k0 = blockIdx.y * 32;
  int tx = threadIdx.x, ty = threadIdx.y;  // 32 x 8
#pragma unroll
  for (int i = 0; i < 32; i += 8)
    t[ty + i][tx] = (__bf16)src[(size_t)(k0 + ty + i) * N + n0 + tx];
  __syncthreads();
#pragma unroll
  for (int i = 0; i < 32; i += 8)
    dst[(size_t)(n0 + ty + i) * K + k0 + tx] = t[tx][ty + i];
}

// ---------------------------------------------------------------------------
// WMMA GEMM: C[M][N] = A[M][K](bf16) * W[K][N]  with W stored as Bt[N][K] bf16.
// Block: 256 thr = 8 waves; block tile 128x128; wave tile 64x32 (4x2 WMMA tiles).
// K staged in 32-deep chunks, double-buffered LDS, TDM-fed when available.
// Epilogues: 0=f32  1=bf16  2=f32+residual(f32)  3=bf16 silu(gate)*acc
// ---------------------------------------------------------------------------
#define LDSA_STRIDE 40  // 32 + pad; 80B rows: 16B aligned, bank-spread

template <int EPI>
__global__ __launch_bounds__(256) void dl_gemm_kernel(
    const __bf16* __restrict__ A, const __bf16* __restrict__ Bt,
    void* __restrict__ out, const void* __restrict__ extra,
    int M, int N, int K) {
  __shared__ __bf16 a_lds[2][128 * LDSA_STRIDE];
  __shared__ __bf16 b_lds[2][128 * LDSA_STRIDE];

  const int blockN = blockIdx.x * 128;
  const int blockM = blockIdx.y * 128;
  const int tid  = threadIdx.x;
  const int wave = tid >> 5;
  const int lane = tid & 31;
  const int wm = wave >> 2;          // 0..1  (64-row slabs)
  const int wn = wave & 3;           // 0..3  (32-col slabs)

  v8f acc[4][2];
#pragma unroll
  for (int mi = 0; mi < 4; ++mi)
#pragma unroll
    for (int ni = 0; ni < 2; ++ni)
      acc[mi][ni] = (v8f){0.f, 0.f, 0.f, 0.f, 0.f, 0.f, 0.f, 0.f};

  const int nk = K / 32;

#ifdef DL_TDM
  // ---- TDM double-buffered pipeline: tile i+1 streams while tile i computes.
  auto issue = [&](int buf, int k0) {
    if (wave == 0) {
      tdm_load_2d((unsigned)(uintptr_t)&a_lds[buf][0],
                  A + (size_t)blockM * K + k0, 32u, 128u,
                  (unsigned)K, (unsigned)M, (unsigned)K);
      tdm_load_2d((unsigned)(uintptr_t)&b_lds[buf][0],
                  Bt + (size_t)blockN * K + k0, 32u, 128u,
                  (unsigned)K, (unsigned)N, (unsigned)K);
    }
  };
  issue(0, 0);
  if (wave == 0) __builtin_amdgcn_s_wait_tensorcnt(0);
  __syncthreads();
  for (int i = 0; i < nk; ++i) {
    const int buf = i & 1;
    if (i + 1 < nk) issue(buf ^ 1, (i + 1) * 32);

    v16bf af[4];
#pragma unroll
    for (int mi = 0; mi < 4; ++mi)
      af[mi] = load_a_frag(a_lds[buf], wm * 64 + mi * 16, 0, LDSA_STRIDE);
    v16bf bfr[2];
#pragma unroll
    for (int ni = 0; ni < 2; ++ni)
      bfr[ni] = load_b_frag(b_lds[buf], wn * 32 + ni * 16, 0, LDSA_STRIDE);
#pragma unroll
    for (int mi = 0; mi < 4; ++mi)
#pragma unroll
      for (int ni = 0; ni < 2; ++ni)
        acc[mi][ni] = wmma_bf16(af[mi], bfr[ni], acc[mi][ni]);

    if (wave == 0 && i + 1 < nk) __builtin_amdgcn_s_wait_tensorcnt(0);
    __syncthreads();
  }
#else
  // ---- Fallback: cooperative vector loads, single buffer.
  const int r  = tid >> 1;
  const int cb = (tid & 1) * 16;
  for (int i = 0; i < nk; ++i) {
    const int k0 = i * 32;
    __syncthreads();
    {
      const uint4* g = reinterpret_cast<const uint4*>(
          A + (size_t)(blockM + r) * K + k0 + cb);
      uint4* l = reinterpret_cast<uint4*>(&a_lds[0][r * LDSA_STRIDE + cb]);
      l[0] = g[0]; l[1] = g[1];
    }
    {
      const uint4* g = reinterpret_cast<const uint4*>(
          Bt + (size_t)(blockN + r) * K + k0 + cb);
      uint4* l = reinterpret_cast<uint4*>(&b_lds[0][r * LDSA_STRIDE + cb]);
      l[0] = g[0]; l[1] = g[1];
    }
    __syncthreads();
    v16bf af[4];
#pragma unroll
    for (int mi = 0; mi < 4; ++mi)
      af[mi] = load_a_frag(a_lds[0], wm * 64 + mi * 16, 0, LDSA_STRIDE);
    v16bf bfr[2];
#pragma unroll
    for (int ni = 0; ni < 2; ++ni)
      bfr[ni] = load_b_frag(b_lds[0], wn * 32 + ni * 16, 0, LDSA_STRIDE);
#pragma unroll
    for (int mi = 0; mi < 4; ++mi)
#pragma unroll
      for (int ni = 0; ni < 2; ++ni)
        acc[mi][ni] = wmma_bf16(af[mi], bfr[ni], acc[mi][ni]);
  }
#endif

  // Epilogue. C layout: vgpr j -> (M=j, N=lane) lanes 0-15 ; (M=j+8, N=lane-16).
  const int hl  = lane >> 4;
  const int cn  = lane & 15;
#pragma unroll
  for (int mi = 0; mi < 4; ++mi)
#pragma unroll
    for (int ni = 0; ni < 2; ++ni)
#pragma unroll
      for (int j = 0; j < 8; ++j) {
        int row = blockM + wm * 64 + mi * 16 + j + 8 * hl;
        int col = blockN + wn * 32 + ni * 16 + cn;
        size_t idx = (size_t)row * N + col;
        float a = acc[mi][ni][j];
        if (EPI == 0) {
          ((float*)out)[idx] = a;
        } else if (EPI == 1) {
          ((__bf16*)out)[idx] = (__bf16)a;
        } else if (EPI == 2) {
          ((float*)out)[idx] = a + ((const float*)extra)[idx];
        } else {  // EPI == 3 : silu(gate) * acc
          float g = (float)((const __bf16*)extra)[idx];
          float s = g / (1.0f + __expf(-g));
          ((__bf16*)out)[idx] = (__bf16)(s * a);
        }
      }
}

// ---------------------------------------------------------------------------
// RoPE + head-major permute: in bf16 [S][nh*64] -> out bf16 [nh][S][64]
// ---------------------------------------------------------------------------
__global__ __launch_bounds__(256) void dl_rope_kernel(
    const __bf16* __restrict__ in, const float* __restrict__ sin_t,
    const float* __restrict__ cos_t, __bf16* __restrict__ out, int nh) {
  int idx = blockIdx.x * blockDim.x + threadIdx.x;
  int total = SQ * nh * 32;
  if (idx >= total) return;
  int d = idx & 31;
  int h = (idx >> 5) % nh;
  int s = idx / (32 * nh);
  size_t base = (size_t)s * (nh * HDIM) + h * HDIM;
  float x0 = (float)in[base + d];
  float x1 = (float)in[base + d + 32];
  float sn = sin_t[s * 32 + d];
  float cs = cos_t[s * 32 + d];
  size_t ob = ((size_t)h * SQ + s) * HDIM;
  out[ob + d]      = (__bf16)(x0 * cs - x1 * sn);
  out[ob + d + 32] = (__bf16)(x0 * sn + x1 * cs);
}

// V permute+transpose: bf16 [S][NKV*64] -> bf16 [NKV][64][S]
__global__ __launch_bounds__(256) void dl_vtrans_kernel(
    const __bf16* __restrict__ in, __bf16* __restrict__ out) {
  int idx = blockIdx.x * blockDim.x + threadIdx.x;
  if (idx >= SQ * NKVH * HDIM) return;
  int c = idx % (NKVH * HDIM);
  int s = idx / (NKVH * HDIM);
  int h = c / HDIM, d = c % HDIM;
  out[((size_t)h * HDIM + d) * SQ + s] = in[idx];
}

// ---------------------------------------------------------------------------
// Flash attention (causal, GQA 4:1).
// Block = 128 thr (4 waves); each wave owns 16 q rows; block = one (head,
// 64-row q block). Scores and P*V on WMMA; online softmax via half-wave
// shuffle row reductions (a C-matrix row lives across one 16-lane half).
// qh: [NH][S][64], kh: [NKV][S][64], vT: [NKV][64][S], attn out: [S][NH*64]
// ---------------------------------------------------------------------------
#define FA_STRIDE 72  // 64 + pad: 144B rows (16B aligned), bank-spread

__global__ __launch_bounds__(128) void dl_flash_kernel(
    const __bf16* __restrict__ qh, const __bf16* __restrict__ kh,
    const __bf16* __restrict__ vT, __bf16* __restrict__ attn) {
  const int qb  = blockIdx.x;        // 64-row q block
  const int h   = blockIdx.y;        // query head
  const int kvh = h >> 2;            // GQA group of 4
  const int tid  = threadIdx.x;
  const int wave = tid >> 5;
  const int lane = tid & 31;
  const int hl   = lane >> 4;
  const int cn   = lane & 15;

  __shared__ __bf16 q_t[64 * FA_STRIDE];
  __shared__ __bf16 k_t[64 * FA_STRIDE];
  __shared__ __bf16 v_t[64 * FA_STRIDE];  // [d][key]
  __shared__ __bf16 p_t[64 * FA_STRIDE];  // per-wave 16-row slabs

  {
    const __bf16* qbase = qh + ((size_t)h * SQ + (size_t)qb * 64) * HDIM;
    for (int e = tid; e < 64 * 32; e += 128) {
      int r = e >> 5, dp = (e & 31) * 2;
      *(unsigned int*)&q_t[r * FA_STRIDE + dp] =
          *(const unsigned int*)&qbase[r * HDIM + dp];
    }
  }

  const float NEG = -1.0e30f;
  float m_s[8], l_s[8];
  v8f O[4];
#pragma unroll
  for (int j = 0; j < 8; ++j) { m_s[j] = NEG; l_s[j] = 0.f; }
#pragma unroll
  for (int ot = 0; ot < 4; ++ot)
    O[ot] = (v8f){0.f, 0.f, 0.f, 0.f, 0.f, 0.f, 0.f, 0.f};

  for (int kb = 0; kb <= qb; ++kb) {
    __syncthreads();  // previous iteration's K/V reads done
    {
      const __bf16* kbase = kh + ((size_t)kvh * SQ + (size_t)kb * 64) * HDIM;
      for (int e = tid; e < 64 * 32; e += 128) {
        int r = e >> 5, dp = (e & 31) * 2;
        *(unsigned int*)&k_t[r * FA_STRIDE + dp] =
            *(const unsigned int*)&kbase[r * HDIM + dp];
      }
      const __bf16* vbase = vT + (size_t)kvh * HDIM * SQ + (size_t)kb * 64;
      for (int e = tid; e < 64 * 32; e += 128) {
        int d = e >> 5, kp = (e & 31) * 2;
        *(unsigned int*)&v_t[d * FA_STRIDE + kp] =
            *(const unsigned int*)&vbase[(size_t)d * SQ + kp];
      }
    }
    __syncthreads();

    // scores S = Q * K^T (scale 1/sqrt(64))
    v8f sc[4];
#pragma unroll
    for (int nt = 0; nt < 4; ++nt) {
      v8f c = (v8f){0.f, 0.f, 0.f, 0.f, 0.f, 0.f, 0.f, 0.f};
#pragma unroll
      for (int kc = 0; kc < 2; ++kc) {
        v16bf a = load_a_frag(q_t, wave * 16, kc * 32, FA_STRIDE);
        v16bf b = load_b_frag(k_t, nt * 16, kc * 32, FA_STRIDE);
        c = wmma_bf16(a, b, c);
      }
      sc[nt] = c;
    }
    const bool diag = (kb == qb);
#pragma unroll
    for (int nt = 0; nt < 4; ++nt)
#pragma unroll
      for (int j = 0; j < 8; ++j) {
        float s = sc[nt][j] * 0.125f;
        if (diag) {
          int rl = wave * 16 + j + 8 * hl;   // row within 64-block
          int cl = nt * 16 + cn;             // col within 64-block
          if (cl > rl) s = NEG;
        }
        sc[nt][j] = s;
      }

    // online softmax
    float nm[8];
#pragma unroll
    for (int j = 0; j < 8; ++j) {
      float v = fmaxf(fmaxf(sc[0][j], sc[1][j]), fmaxf(sc[2][j], sc[3][j]));
      v = half_reduce_max(v);
      nm[j] = fmaxf(m_s[j], v);
    }
#pragma unroll
    for (int nt = 0; nt < 4; ++nt)
#pragma unroll
      for (int j = 0; j < 8; ++j) sc[nt][j] = __expf(sc[nt][j] - nm[j]);
#pragma unroll
    for (int j = 0; j < 8; ++j) {
      float r = sc[0][j] + sc[1][j] + sc[2][j] + sc[3][j];
      r = half_reduce_sum(r);
      float alpha = __expf(m_s[j] - nm[j]);
      l_s[j] = l_s[j] * alpha + r;
      m_s[j] = nm[j];
#pragma unroll
      for (int ot = 0; ot < 4; ++ot) O[ot][j] = O[ot][j] * alpha;
    }

    // stage P (bf16) into this wave's LDS slab, then O += P * V
#pragma unroll
    for (int nt = 0; nt < 4; ++nt)
#pragma unroll
      for (int j = 0; j < 8; ++j)
        p_t[(wave * 16 + j + 8 * hl) * FA_STRIDE + nt * 16 + cn] =
            (__bf16)sc[nt][j];
    // wave-local LDS RAW: DS ops from the same wave stay in order
#pragma unroll
    for (int ot = 0; ot < 4; ++ot)
#pragma unroll
      for (int kc = 0; kc < 2; ++kc) {
        v16bf a = load_a_frag(p_t, wave * 16, kc * 32, FA_STRIDE);
        v16bf b = load_b_frag(v_t, ot * 16, kc * 32, FA_STRIDE);
        O[ot] = wmma_bf16(a, b, O[ot]);
      }
  }

  // finalize: O/l -> attn[S][NH*64]
#pragma unroll
  for (int j = 0; j < 8; ++j) {
    float invl = 1.0f / l_s[j];
    int row = qb * 64 + wave * 16 + j + 8 * hl;
#pragma unroll
    for (int ot = 0; ot < 4; ++ot) {
      int col = h * HDIM + ot * 16 + cn;
      attn[(size_t)row * (NHQ * HDIM) + col] = (__bf16)(O[ot][j] * invl);
    }
  }
}

// ---------------------------------------------------------------------------
// Host driver
// ---------------------------------------------------------------------------
extern "C" void kernel_launch(void* const* d_in, const int* in_sizes, int n_in,
                              void* d_out, int out_size, void* d_ws,
                              size_t ws_size, hipStream_t stream) {
  (void)in_sizes; (void)n_in; (void)out_size; (void)ws_size;
  const float* hidden = (const float*)d_in[0];
  const float* sin_t  = (const float*)d_in[1];
  const float* cos_t  = (const float*)d_in[2];
  const float* ln1    = (const float*)d_in[3];
  const float* ln2    = (const float*)d_in[4];
  const float* wq     = (const float*)d_in[5];
  const float* wk     = (const float*)d_in[6];
  const float* wv     = (const float*)d_in[7];
  const float* wo     = (const float*)d_in[8];
  const float* wg     = (const float*)d_in[9];
  const float* wu     = (const float*)d_in[10];
  const float* wd     = (const float*)d_in[11];
  float* out = (float*)d_out;  // also serves as x2 residual buffer

  char* p = (char*)d_ws;
  auto alloc = [&](size_t bytes) {
    char* r = p;
    p += (bytes + 255) & ~(size_t)255;
    return r;
  };
  __bf16* wT   = (__bf16*)alloc((size_t)INTR * HIDD * 2);  // reused per weight
  __bf16* xn   = (__bf16*)alloc((size_t)SQ * HIDD * 2);
  __bf16* qbuf = (__bf16*)alloc((size_t)SQ * NHQ * HDIM * 2);
  __bf16* kbuf = (__bf16*)alloc((size_t)SQ * NKVH * HDIM * 2);
  __bf16* vbuf = (__bf16*)alloc((size_t)SQ * NKVH * HDIM * 2);
  __bf16* qhh  = (__bf16*)alloc((size_t)NHQ * SQ * HDIM * 2);
  __bf16* khh  = (__bf16*)alloc((size_t)NKVH * SQ * HDIM * 2);
  __bf16* vTb  = (__bf16*)alloc((size_t)NKVH * HDIM * SQ * 2);
  __bf16* attn = (__bf16*)alloc((size_t)SQ * NHQ * HDIM * 2);
  __bf16* hb   = (__bf16*)alloc((size_t)SQ * HIDD * 2);
  __bf16* gate = (__bf16*)alloc((size_t)SQ * INTR * 2);
  __bf16* act  = (__bf16*)alloc((size_t)SQ * INTR * 2);

  dim3 cblk(32, 8);
  auto convt = [&](const float* src, int K, int N) {
    dl_convt_kernel<<<dim3(N / 32, K / 32), cblk, 0, stream>>>(src, wT, K, N);
  };
  auto gemm_grid = [](int M, int N) { return dim3(N / 128, M / 128); };

  // 1) rmsnorm1
  dl_rmsnorm_kernel<<<SQ, 256, 0, stream>>>(hidden, ln1, xn);

  // 2) Q/K/V projections (bf16 out)
  convt(wq, HIDD, NHQ * HDIM);
  dl_gemm_kernel<1><<<gemm_grid(SQ, NHQ * HDIM), 256, 0, stream>>>(
      xn, wT, qbuf, nullptr, SQ, NHQ * HDIM, HIDD);
  convt(wk, HIDD, NKVH * HDIM);
  dl_gemm_kernel<1><<<gemm_grid(SQ, NKVH * HDIM), 256, 0, stream>>>(
      xn, wT, kbuf, nullptr, SQ, NKVH * HDIM, HIDD);
  convt(wv, HIDD, NKVH * HDIM);
  dl_gemm_kernel<1><<<gemm_grid(SQ, NKVH * HDIM), 256, 0, stream>>>(
      xn, wT, vbuf, nullptr, SQ, NKVH * HDIM, HIDD);

  // 3) RoPE + layout permutes
  dl_rope_kernel<<<(SQ * NHQ * 32 + 255) / 256, 256, 0, stream>>>(
      qbuf, sin_t, cos_t, qhh, NHQ);
  dl_rope_kernel<<<(SQ * NKVH * 32 + 255) / 256, 256, 0, stream>>>(
      kbuf, sin_t, cos_t, khh, NKVH);
  dl_vtrans_kernel<<<(SQ * NKVH * HDIM + 255) / 256, 256, 0, stream>>>(vbuf, vTb);

  // 4) causal flash attention
  dl_flash_kernel<<<dim3(SQ / 64, NHQ), 128, 0, stream>>>(qhh, khh, vTb, attn);

  // 5) O projection + residual -> x2 (stored in d_out)
  convt(wo, NHQ * HDIM, HIDD);
  dl_gemm_kernel<2><<<gemm_grid(SQ, HIDD), 256, 0, stream>>>(
      attn, wT, out, hidden, SQ, HIDD, NHQ * HDIM);

  // 6) rmsnorm2
  dl_rmsnorm_kernel<<<SQ, 256, 0, stream>>>(out, ln2, hb);

  // 7) MLP: gate, silu(gate)*up, down + residual
  convt(wg, HIDD, INTR);
  dl_gemm_kernel<1><<<gemm_grid(SQ, INTR), 256, 0, stream>>>(
      hb, wT, gate, nullptr, SQ, INTR, HIDD);
  convt(wu, HIDD, INTR);
  dl_gemm_kernel<3><<<gemm_grid(SQ, INTR), 256, 0, stream>>>(
      hb, wT, act, gate, SQ, INTR, HIDD);
  convt(wd, INTR, HIDD);
  dl_gemm_kernel<2><<<gemm_grid(SQ, HIDD), 256, 0, stream>>>(
      act, wT, out, out, SQ, HIDD, INTR);
}
